// Decoder_57758720196948
// MI455X (gfx1250) — matrix operand order repaired and checked by reading, not compiled, not run on hardware
//
#include <hip/hip_runtime.h>
#include <hip/hip_bf16.h>
#include <math.h>

// ---------------------------------------------------------------------------
// Decoder step for MI455X (gfx1250, wave32, WMMA).
// All GEMMs via v_wmma_f32_16x16x32_bf16 (fp32->bf16 via v_cvt_pk_bf16_f32 in
// the LDS producers, fp32 accumulation). Score tensor fused away (tanh + dot-v
// in GEMM epilogue). gru_rk GEMM eliminated (h_prev == 0 in reference).
// Double-buffered LDS (1 barrier / K-step), global->register staging pipelined
// one K-tile ahead; each wave computes a 32x32 tile (2x2 fragment reuse).
// ---------------------------------------------------------------------------

typedef __attribute__((ext_vector_type(16))) __bf16 v16bf;
typedef __attribute__((ext_vector_type(2)))  __bf16 v2bf;
typedef __attribute__((ext_vector_type(8)))  float  v8f;

union BF16Frag { unsigned u[8]; v16bf v; };

// pack two floats to packed bf16 in one v_cvt_pk_bf16_f32
__device__ __forceinline__ unsigned pk2bf(float x, float y) {
#if __has_builtin(__builtin_amdgcn_cvt_pk_bf16_f32)
  return __builtin_bit_cast(unsigned, __builtin_amdgcn_cvt_pk_bf16_f32(x, y));
#else
  v2bf p;
  p[0] = (__bf16)x;
  p[1] = (__bf16)y;
  return __builtin_bit_cast(unsigned, p);
#endif
}

#define BM 64
#define BN 64
#define BK 32
#define GEMM_THREADS 128

// mode 0: out[m,n] = acc + bias[n]                  (guarded for odd N)
// mode 1: e[m] += sum_n tanh(acc + bias[n] + hb[m>>6, n]) * vvec[n]
__global__ __launch_bounds__(GEMM_THREADS)
void gemm_wmma_bf16(const float* __restrict__ A, const float* __restrict__ B,
                    const float* __restrict__ bias, float* __restrict__ out,
                    int M, int N, int K, int mode,
                    const float* __restrict__ hb, const float* __restrict__ vvec,
                    float* __restrict__ e)
{
  __shared__ __bf16 As[2][BM][BK];   // M x K (row major), double buffered
  __shared__ __bf16 Bt[2][BN][BK];   // N x K (transposed), double buffered

  const int t    = threadIdx.x;
  const int lane = t & 31;
  const int wave = t >> 5;
  const int mq   = wave >> 1;       // wave's M quadrant (0/1) -> rows 32*mq..
  const int nq   = wave & 1;        // wave's N quadrant (0/1) -> cols 32*nq..
  const int n0   = blockIdx.x * BN;
  const int m0   = blockIdx.y * BM;

  v8f acc[2][2] = {};

  const int arow = t >> 3;          // 0..15  (A loader: 8 thr/row * 4 floats)
  const int acol = (t & 7) * 4;     // 0..28
  const int bnc  = (t & 15) * 4;    // 0..60  (B loader: 4 n cols ...)
  const int bk0  = (t >> 4) * 4;    //        ... x 4 consecutive k rows

  const bool nAligned = (N & 3) == 0;   // 1024/3072 fast path; 50257 guarded
  const int  hi = lane >> 4;            // lane half (K-split in fragments)
  const int  ml = lane & 15;

  float4 stA[4];
  float  stB[4][4];                 // [k-sub 0..3][n-sub 0..3]

  auto loadTile = [&](int kk) {
    #pragma unroll
    for (int p = 0; p < 4; ++p) {
      const int r = p * 16 + arow;
      stA[p] = *(const float4*)&A[(size_t)(m0 + r) * K + kk + acol];
    }
    #pragma unroll
    for (int p = 0; p < 4; ++p) {   // 4 consecutive k rows, same n cols
      const float* src = &B[(size_t)(kk + bk0 + p) * N + n0 + bnc];
      if (nAligned) {
        const float4 f = *(const float4*)src;
        stB[p][0] = f.x; stB[p][1] = f.y; stB[p][2] = f.z; stB[p][3] = f.w;
      } else {
        #pragma unroll
        for (int j = 0; j < 4; ++j)
          stB[p][j] = (n0 + bnc + j < N) ? src[j] : 0.f;
      }
    }
  };

  auto storeTile = [&](int buf) {
    #pragma unroll
    for (int p = 0; p < 4; ++p) {
      const int r = p * 16 + arow;
      uint2 w;
      w.x = pk2bf(stA[p].x, stA[p].y);
      w.y = pk2bf(stA[p].z, stA[p].w);
      *(uint2*)&As[buf][r][acol] = w;        // 8B aligned (acol % 4 == 0)
    }
    #pragma unroll
    for (int j = 0; j < 4; ++j) {            // per n col: 4 k's -> one b64
      uint2 w;
      w.x = pk2bf(stB[0][j], stB[1][j]);     // (k0, k0+1)
      w.y = pk2bf(stB[2][j], stB[3][j]);     // (k0+2, k0+3)
      *(uint2*)&Bt[buf][bnc + j][bk0] = w;   // 8B aligned (bk0 % 4 == 0)
    }
  };

  loadTile(0);
  storeTile(0);
  if (BK < K) loadTile(BK);

  int it = 0;
  for (int kk = 0; kk < K; kk += BK, ++it) {
    __syncthreads();                 // buf[it&1] is fully written
    const int cur = it & 1;

    // ---- A fragments (ISA 7.12.2, 16-bit A 16x32):
    //   lane<16: VGPR j<4 -> K=2j,2j+1 ; VGPR 4..7 -> K=16+2(j-4)
    //   lane>=16: +8 on K
    BF16Frag fa[2];
    #pragma unroll
    for (int mt = 0; mt < 2; ++mt) {
      const unsigned short* ap =
          (const unsigned short*)&As[cur][mq * 32 + mt * 16 + ml][0];
      #pragma unroll
      for (int j = 0; j < 8; ++j) {
        const int k0 = ((j < 4) ? 2 * j : 16 + 2 * (j - 4)) + hi * 8;
        fa[mt].u[j] = *(const unsigned*)&ap[k0];
      }
    }
    // ---- B fragments: same K layout as A (lane>=16: +8 on K), pairs/VGPR ----
    BF16Frag fb[2];
    #pragma unroll
    for (int nt = 0; nt < 2; ++nt) {
      const unsigned short* bp =
          (const unsigned short*)&Bt[cur][nq * 32 + nt * 16 + ml][0];
      #pragma unroll
      for (int j = 0; j < 8; ++j)
        fb[nt].u[j] = *(const unsigned*)&bp[((j < 4) ? 2 * j : 16 + 2 * (j - 4)) + hi * 8];
    }

    // ---- stage tile kk+BK into the other buffer; fetch tile kk+2*BK -------
    if (kk + BK < K) {
      storeTile(1 - cur);
      if (kk + 2 * BK < K) {   // prefetch tile after next (global_prefetch_b8)
        __builtin_prefetch(&A[(size_t)(m0 + arow) * K + kk + 2 * BK + acol], 0, 1);
        __builtin_prefetch(&B[(size_t)(kk + 2 * BK + bk0) * N + n0 + bnc], 0, 1);
        loadTile(kk + 2 * BK);
      }
    }

    // ---- 2x2 outer product of fragments -----------------------------------
    #pragma unroll
    for (int mt = 0; mt < 2; ++mt)
      #pragma unroll
      for (int nt = 0; nt < 2; ++nt)
        acc[mt][nt] = __builtin_amdgcn_wmma_f32_16x16x32_bf16(
            false, fa[mt].v, false, fb[nt].v, (short)0, acc[mt][nt],
            false, false);
  }

  // ---- epilogue -----------------------------------------------------------
  // C/D layout: VGPR r, lane<16: (M=r, N=lane); lane>=16: (M=8+r, N=lane-16)
  if (mode == 0) {
    #pragma unroll
    for (int mt = 0; mt < 2; ++mt) {
      const int mbase = m0 + mq * 32 + mt * 16 + hi * 8;
      #pragma unroll
      for (int nt = 0; nt < 2; ++nt) {
        const int n = n0 + nq * 32 + nt * 16 + ml;
        if (n < N) {
          const float bb = bias ? bias[n] : 0.f;
          #pragma unroll
          for (int r = 0; r < 8; ++r)
            out[(size_t)(mbase + r) * N + n] = acc[mt][nt][r] + bb;
        }
      }
    }
  } else {
    // score-GEMM fusion: val = tanh(acc + b1[n] + hb[b,n]); e[m] += val*v[n]
    float vv[2], bb[2];
    const int b = m0 >> 6;                 // BM==L==64 -> whole block same b
    const float* hbrow = &hb[(size_t)b * N];
    #pragma unroll
    for (int nt = 0; nt < 2; ++nt) {
      const int n = n0 + nq * 32 + nt * 16 + ml;
      vv[nt] = vvec[n];
      bb[nt] = bias[n] + hbrow[n];
    }
    #pragma unroll
    for (int mt = 0; mt < 2; ++mt) {
      const int mbase = m0 + mq * 32 + mt * 16 + hi * 8;
      #pragma unroll
      for (int r = 0; r < 8; ++r) {
        float s = 0.f;
        #pragma unroll
        for (int nt = 0; nt < 2; ++nt)
          s += tanhf(acc[mt][nt][r] + bb[nt]) * vv[nt];
        // reduce across the 16 lanes of each half (rows differ per half)
        #pragma unroll
        for (int off = 8; off >= 1; off >>= 1) s += __shfl_xor(s, off, 32);
        if (ml == 0) atomicAdd(&e[mbase + r], s);   // lane 0 and lane 16
      }
    }
  }
}

// ---------------------------------------------------------------------------
__global__ void zero_f32(float* p, int n) {
  int i = blockIdx.x * blockDim.x + threadIdx.x;
  if (i < n) p[i] = 0.f;
}

// softmax over L=64 per batch row; one wave32 per row. (bv is a constant
// added to every logit -> cancels exactly in softmax, omitted.)
__global__ __launch_bounds__(32) void softmax64(const float* __restrict__ e,
                                                float* __restrict__ w) {
  const int b = blockIdx.x, t = threadIdx.x;
  float a0 = e[b * 64 + t], a1 = e[b * 64 + 32 + t];
  float m = fmaxf(a0, a1);
  #pragma unroll
  for (int s = 16; s >= 1; s >>= 1) m = fmaxf(m, __shfl_xor(m, s, 32));
  float e0 = expf(a0 - m), e1 = expf(a1 - m);
  float sum = e0 + e1;
  #pragma unroll
  for (int s = 16; s >= 1; s >>= 1) sum += __shfl_xor(sum, s, 32);
  const float inv = 1.f / sum;
  w[b * 64 + t] = e0 * inv;
  w[b * 64 + 32 + t] = e1 * inv;
}

// contextVec (weighted sum over L) + embedding gather -> x = [ctx | emb[X]]
__global__ __launch_bounds__(256)
void context_concat(const float* __restrict__ enc, const float* __restrict__ attn,
                    const int* __restrict__ X, const float* __restrict__ emb,
                    float* __restrict__ x) {
  const int b = blockIdx.x, t = threadIdx.x;
  __shared__ float w[64];
  if (t < 64) w[t] = attn[b * 64 + t];
  __syncthreads();
  float acc[8] = {};
  const float* eb = enc + (size_t)b * 64 * 2048;
  for (int l = 0; l < 64; ++l) {
    const float wl = w[l];
    const float* er = eb + l * 2048;
    #pragma unroll
    for (int i = 0; i < 8; ++i) acc[i] += wl * er[t + 256 * i];
  }
  float* xb = x + (size_t)b * 2560;
  #pragma unroll
  for (int i = 0; i < 8; ++i) xb[t + 256 * i] = acc[i];
  const float* em = emb + (size_t)X[b] * 512;
  #pragma unroll
  for (int i = 0; i < 2; ++i) xb[2048 + t + 256 * i] = em[t + 256 * i];
}

// GRU pointwise with h_prev==0: hm = gru_b[1]; finalState = (1-z)*cand
__global__ __launch_bounds__(256)
void gru_combine(const float* __restrict__ xm, const float* __restrict__ gru_b,
                 float* __restrict__ fs) {
  const int i = blockIdx.x * blockDim.x + threadIdx.x;   // 256*1024 total
  const int b = i >> 10, u = i & 1023;
  const float* gb1 = gru_b + 3072;
  const float xz = xm[(size_t)b * 3072 + u]          + gb1[u];
  const float xr = xm[(size_t)b * 3072 + 1024 + u]   + gb1[1024 + u];
  const float xh = xm[(size_t)b * 3072 + 2048 + u];
  const float z  = 1.f / (1.f + expf(-xz));
  const float r  = 1.f / (1.f + expf(-xr));
  const float c  = tanhf(xh + r * gb1[2048 + u]);
  fs[i] = (1.f - z) * c;
}

// ---------------------------------------------------------------------------
extern "C" void kernel_launch(void* const* d_in, const int* in_sizes, int n_in,
                              void* d_out, int out_size, void* d_ws, size_t ws_size,
                              hipStream_t stream) {
  (void)in_sizes; (void)n_in; (void)out_size; (void)ws_size;

  const int*   X      = (const int*)  d_in[0];
  const float* enc    = (const float*)d_in[1];
  const float* hidden = (const float*)d_in[2];
  const float* emb    = (const float*)d_in[3];
  const float* W1     = (const float*)d_in[4];
  const float* b1     = (const float*)d_in[5];
  const float* W2     = (const float*)d_in[6];
  const float* b2     = (const float*)d_in[7];
  const float* v      = (const float*)d_in[8];
  /* d_in[9] = bv: softmax-invariant constant, omitted */
  const float* gru_k  = (const float*)d_in[10];
  /* d_in[11] = gru_rk: multiplied by h_prev==0, dead */
  const float* gru_b  = (const float*)d_in[12];
  const float* fc1_w  = (const float*)d_in[13];
  const float* fc1_b  = (const float*)d_in[14];
  const float* fc2_w  = (const float*)d_in[15];
  const float* fc2_b  = (const float*)d_in[16];

  // workspace layout (floats): hb | e | x | xm | o1  (~7.9 MB)
  float* ws = (float*)d_ws;
  float* hb = ws;                    // 256*1024
  float* e  = hb + 256 * 1024;       // 16384
  float* x  = e  + 16384;            // 256*2560
  float* xm = x  + 256 * 2560;       // 256*3072
  float* o1 = xm + 256 * 3072;       // 256*1024

  // d_out layout: logits (256*50257) | finalState (256*1024) | attn (256*64)
  float* out    = (float*)d_out;
  float* logits = out;
  float* fs     = out + (size_t)256 * 50257;
  float* attn   = fs  + 256 * 1024;

  // 1) e = 0
  zero_f32<<<64, 256, 0, stream>>>(e, 16384);

  // 2) hb = hidden @ W2 + b2                     (256 x 1024, K=1024)
  gemm_wmma_bf16<<<dim3(16, 4), GEMM_THREADS, 0, stream>>>(
      hidden, W2, b2, hb, 256, 1024, 1024, 0, nullptr, nullptr, nullptr);

  // 3) fused score GEMM: e[b,l] = sum_u tanh(enc@W1 + b1 + hb)*v   (68.7 GF)
  gemm_wmma_bf16<<<dim3(16, 256), GEMM_THREADS, 0, stream>>>(
      enc, W1, b1, nullptr, 16384, 1024, 2048, 1, hb, v, e);

  // 4) attnWts = softmax_L(e)
  softmax64<<<256, 32, 0, stream>>>(e, attn);

  // 5) x = [sum_l attn*enc | emb[X]]
  context_concat<<<256, 256, 0, stream>>>(enc, attn, X, emb, x);

  // 6) xm = x @ gru_k + gru_b[0]                 (256 x 3072, K=2560)
  gemm_wmma_bf16<<<dim3(48, 4), GEMM_THREADS, 0, stream>>>(
      x, gru_k, gru_b, xm, 256, 3072, 2560, 0, nullptr, nullptr, nullptr);

  // 7) finalState = (1-z)*cand
  gru_combine<<<1024, 256, 0, stream>>>(xm, gru_b, fs);

  // 8) o1 = finalState @ fc1_w + fc1_b           (256 x 1024, K=1024)
  gemm_wmma_bf16<<<dim3(16, 4), GEMM_THREADS, 0, stream>>>(
      fs, fc1_w, fc1_b, o1, 256, 1024, 1024, 0, nullptr, nullptr, nullptr);

  // 9) logits = o1 @ fc2_w + fc2_b               (256 x 50257, K=1024)
  gemm_wmma_bf16<<<dim3(786, 4), GEMM_THREADS, 0, stream>>>(
      o1, fc2_w, fc2_b, logits, 256, 50257, 1024, 0, nullptr, nullptr, nullptr);
}
